// CoattentionModel_37151467110636
// MI455X (gfx1250) — matrix operand order repaired and checked by reading, not compiled, not run on hardware
//
#include <hip/hip_runtime.h>

// ---------------------------------------------------------------------------
// CDNA5 (gfx1250) co-attention pipeline.
// All GEMM stages on v_wmma_f32_16x16x32_bf16 (f32 accumulate); A tiles fed by
// the Tensor Data Mover (tensor_load_to_lds + s_wait_tensorcnt, double
// buffered), B tiles by vectorized b128 loads packed in VGPRs and committed
// with ds_store_b128. Workgroup tile 64x256, wave tile 32x64 (8 WMMA/K-step).
// Roofline: ~360 GFLOP vs ~230MB traffic -> matrix-core bound -> bf16 WMMA.
// ---------------------------------------------------------------------------

typedef __bf16 bf16;
typedef __attribute__((ext_vector_type(16))) __bf16    v16bf;
typedef __attribute__((ext_vector_type(8)))  float     v8f;
typedef __attribute__((ext_vector_type(4)))  unsigned  su4;
typedef __attribute__((ext_vector_type(8)))  unsigned  su8;

union FragBF { v16bf v; unsigned u[8]; };

__device__ __forceinline__ v8f wmma_bf16(const FragBF& a, const FragBF& b, v8f c) {
  // (neg_a, A, neg_b, B, c_mod, C, reuse_a, reuse_b)
  return __builtin_amdgcn_wmma_f32_16x16x32_bf16(false, a.v, false, b.v, (short)0, c, false, false);
}

// ---------------------------------------------------------------------------
// TDM: load a 64-row x 32-elem bf16 tile (row stride = rowStride elems) from
// global into LDS at ldsAddr. LDS padding: 16 bytes after every 64-byte row
// (pad_interval=16 dwords, pad_amount=4 dwords) -> 80B LDS row stride, the
// conflict-free layout the WMMA A-fragment reader uses.
// Descriptor layout per CDNA5 ISA ch.8 (D# groups 0..3).
// ---------------------------------------------------------------------------
__device__ __forceinline__ void tdm_load_tile_a(const bf16* gsrc, unsigned ldsAddr,
                                                unsigned rowLen, unsigned long long rowStride)
{
  unsigned long long ga = (unsigned long long)gsrc;
  su4 g0; su8 g1; su4 g2; su4 g3;
  g0[0] = 1u;                                   // count=1, user mode, no gather
  g0[1] = ldsAddr;                              // lds_addr
  g0[2] = (unsigned)ga;                         // global_addr[31:0]
  g0[3] = (unsigned)(ga >> 32) | (2u << 30);    // global_addr[56:32] | type=2
  g1[0] = (1u << 16)                            // data_size = 1 (2 bytes)
        | (1u << 20)                            // pad_enable
        | (3u << 22)                            // pad_interval: 16 dwords
        | (3u << 25);                           // pad_amount: 4 dwords (16B)
  g1[1] = (rowLen & 0xffffu) << 16;             // tensor_dim0[15:0]
  g1[2] = (rowLen >> 16) | (64u << 16);         // tensor_dim0[31:16] | tensor_dim1=64
  g1[3] = (32u << 16);                          // tensor_dim1 hi=0 | tile_dim0=32
  g1[4] = 64u;                                  // tile_dim1=64, tile_dim2=0
  g1[5] = (unsigned)(rowStride & 0xffffffffu);  // tensor_dim0_stride lo
  g1[6] = (unsigned)((rowStride >> 32) & 0xffffu);
  g1[7] = 0u;
  g2[0] = 0u; g2[1] = 0u; g2[2] = 0u; g2[3] = 0u;
  g3[0] = 0u; g3[1] = 0u; g3[2] = 0u; g3[3] = 0u;
  asm volatile("tensor_load_to_lds %0, %1, %2, %3"
               :: "s"(g0), "s"(g1), "s"(g2), "s"(g3)
               : "memory");
}

__device__ __forceinline__ void pack8(uint4 q0, uint4 q1, unsigned* out) {
  out[0] = (q0.x & 0xffffu) | (q1.x << 16);  out[1] = (q0.x >> 16) | (q1.x & 0xffff0000u);
  out[2] = (q0.y & 0xffffu) | (q1.y << 16);  out[3] = (q0.y >> 16) | (q1.y & 0xffff0000u);
  out[4] = (q0.z & 0xffffu) | (q1.z << 16);  out[5] = (q0.z >> 16) | (q1.z & 0xffff0000u);
  out[6] = (q0.w & 0xffffu) | (q1.w << 16);  out[7] = (q0.w >> 16) | (q1.w & 0xffff0000u);
}

// B tile loader: thread owns K-pair rows (2p, 2p+1) x 16 columns; four b128
// global loads (fast path) packed into {lo,hi} bf16 dwords; branch-free
// clamp+select on the N edge.
__device__ __forceinline__ void load_b_pack(const bf16* __restrict__ Bb, int N, int k0,
                                            int bp, int bseg, int n0, unsigned out[16])
{
  const unsigned short* r0 = reinterpret_cast<const unsigned short*>(
      Bb + (long long)(k0 + 2 * bp) * N);
  const int nb = n0 + bseg * 16;
  if (n0 + 256 <= N) {
    const uint4* p0 = reinterpret_cast<const uint4*>(r0 + nb);
    const uint4* p1 = reinterpret_cast<const uint4*>(r0 + N + nb);
    pack8(p0[0], p1[0], out);
    pack8(p0[1], p1[1], out + 8);
  } else {
    #pragma unroll
    for (int j = 0; j < 16; ++j) {
      int n  = nb + j;
      int nc = n < N ? n : N - 1;            // clamp -> unconditional load
      unsigned lo = r0[nc];
      unsigned hi = r0[N + nc];
      if (n >= N) { lo = 0u; hi = 0u; }      // cndmask, no branch
      out[j] = lo | (hi << 16);
    }
  }
}

// ------------------------------ conversions --------------------------------

__global__ __launch_bounds__(256)
void cvt_f32_bf16(const float* __restrict__ in, bf16* __restrict__ out, long long n) {
  long long i = (long long)blockIdx.x * 256 + threadIdx.x;
  if (i < n) out[i] = (bf16)in[i];
}

// w: [512][512][3][3] (OIHW) -> wr: [9][512][512] (tap-major, contiguous K rows)
__global__ __launch_bounds__(256)
void reorder_w3x3(const float* __restrict__ w, bf16* __restrict__ wr) {
  long long i = (long long)blockIdx.x * 256 + threadIdx.x;
  const long long total = 512LL * 512 * 9;
  if (i >= total) return;
  int tap = (int)(i % 9);
  long long cc = i / 9;  // co*512 + ci
  wr[(long long)tap * 512 * 512 + cc] = (bf16)w[i];
}

// ------------------------- generic WMMA bf16 GEMM --------------------------
// C[M,N] = A[M,K] * B[K,N], row-major bf16, f32 accumulate.
// OUT_MODE: 0 -> f32 C[row*ldc+col]; 1 -> bf16 C[row*ldc+col]; 2 -> bf16 C[col*ldc+row]
// Requirements: M % 64 == 0, K % 32 == 0 (true at all call sites); N edge guarded.

template<int OUT_MODE>
__global__ __launch_bounds__(256)
void gemm_bf16_wmma(const bf16* __restrict__ Ag, const bf16* __restrict__ Bg,
                    void* __restrict__ Cp,
                    int M, int N, int K,
                    long long sA, long long sB, long long sC, int ldc)
{
  __shared__ bf16 As[2][64][40];     // double-buffered TDM target (80B row stride)
  __shared__ unsigned Bs[16][260];   // [kpair][n]: {lo=K even, hi=K odd} bf16 pair

  const int bz = blockIdx.z;
  const bf16* Ab = Ag + (long long)bz * sA;
  const bf16* Bb = Bg + (long long)bz * sB;

  const int m0 = blockIdx.y * 64;
  const int n0 = blockIdx.x * 256;
  const int t    = threadIdx.x;
  const int lane = t & 31;           // wave32
  const int wave = t >> 5;           // 8 waves: 2 (M) x 4 (N)
  const int wm   = (wave >> 2) * 32;
  const int wn   = (wave & 3) * 64;
  const int half = lane >> 4;
  const int l16  = lane & 15;
  const int bp   = t >> 4;           // 0..15 K-pair
  const int bseg = t & 15;           // 0..15 column 16-chunk

  v8f c0[4], c1[4];
  #pragma unroll
  for (int j = 0; j < 4; ++j) { v8f z = {}; c0[j] = z; c1[j] = z; }
  unsigned pb[16];

  const int steps = K >> 5;

  // prologue: TDM A(k=0) in flight, B(k=0) in registers
  if (wave == 0)
    tdm_load_tile_a(Ab + (long long)m0 * K,
                    (unsigned)(unsigned long long)&As[0][0][0],
                    (unsigned)K, (unsigned long long)K);
  load_b_pack(Bb, N, 0, bp, bseg, n0, pb);

  for (int i = 0; i < steps; ++i) {
    const int cur = i & 1;
    __syncthreads();   // previous iteration's fragment reads complete
    {                  // commit B registers -> LDS (four ds_store_b128)
      uint4* dst = reinterpret_cast<uint4*>(&Bs[bp][bseg * 16]);
      dst[0] = make_uint4(pb[0],  pb[1],  pb[2],  pb[3]);
      dst[1] = make_uint4(pb[4],  pb[5],  pb[6],  pb[7]);
      dst[2] = make_uint4(pb[8],  pb[9],  pb[10], pb[11]);
      dst[3] = make_uint4(pb[12], pb[13], pb[14], pb[15]);
    }
    if (i + 1 < steps) {  // prefetch next: TDM A -> other buffer, B -> regs
      const int k1 = (i + 1) * 32;
      if (wave == 0)
        tdm_load_tile_a(Ab + (long long)m0 * K + k1,
                        (unsigned)(unsigned long long)&As[cur ^ 1][0][0],
                        (unsigned)K, (unsigned long long)K);
      load_b_pack(Bb, N, k1, bp, bseg, n0, pb);
      if (wave == 0) __builtin_amdgcn_s_wait_tensorcnt(1);  // current tile landed
    } else {
      if (wave == 0) __builtin_amdgcn_s_wait_tensorcnt(0);
    }
    __syncthreads();

    // fragments per documented CDNA5 16-bit layouts
    FragBF a0, a1, bf[4];
    #pragma unroll
    for (int v = 0; v < 8; ++v) {
      int kk = (v < 4 ? 2 * v : 16 + 2 * (v - 4)) + 8 * half;    // A 16x32
      a0.u[v] = *reinterpret_cast<const unsigned*>(&As[cur][wm + l16][kk]);
      a1.u[v] = *reinterpret_cast<const unsigned*>(&As[cur][wm + 16 + l16][kk]);
      int p = (v < 4 ? v : v + 4) + 4 * half;                     // B 32x16
      #pragma unroll
      for (int tj = 0; tj < 4; ++tj)
        bf[tj].u[v] = Bs[p][wn + tj * 16 + l16];
    }
    #pragma unroll
    for (int tj = 0; tj < 4; ++tj) {
      c0[tj] = wmma_bf16(a0, bf[tj], c0[tj]);
      c1[tj] = wmma_bf16(a1, bf[tj], c1[tj]);
    }
  }

  // store per C/D layout: lane N=l16, VGPR r -> M = r + 8*half
  float* Cf = (float*)Cp;
  bf16*  Ch = (bf16*)Cp;
  const long long cb = (long long)bz * sC;
  #pragma unroll
  for (int ti = 0; ti < 2; ++ti) {
    #pragma unroll
    for (int tj = 0; tj < 4; ++tj) {
      v8f acc = ti == 0 ? c0[tj] : c1[tj];
      int col = n0 + wn + tj * 16 + l16;
      bool colOK = col < N;
      #pragma unroll
      for (int r = 0; r < 8; ++r) {
        int row = m0 + wm + ti * 16 + 8 * half + r;
        float val = acc[r];
        if (colOK) {
          if      (OUT_MODE == 0) Cf[cb + (long long)row * ldc + col] = val;
          else if (OUT_MODE == 1) Ch[cb + (long long)row * ldc + col] = (bf16)val;
          else                    Ch[cb + (long long)col * ldc + row] = (bf16)val;
        }
      }
    }
  }
}

// ----------------------------- softmax stages ------------------------------

__global__ __launch_bounds__(256)
void row_stats_kernel(const float* __restrict__ A, float* __restrict__ rmax,
                      float* __restrict__ rsum, int N)
{
  __shared__ float red[256];
  const int row = blockIdx.x, b = blockIdx.y, t = threadIdx.x;
  const float* Ar = A + ((long long)b * N + row) * N;
  float m = -3.4e38f;
  for (int i = t; i < N; i += 256) m = fmaxf(m, Ar[i]);
  red[t] = m; __syncthreads();
  for (int s = 128; s > 0; s >>= 1) { if (t < s) red[t] = fmaxf(red[t], red[t + s]); __syncthreads(); }
  m = red[0]; __syncthreads();
  float sum = 0.f;
  for (int i = t; i < N; i += 256) sum += __expf(Ar[i] - m);
  red[t] = sum; __syncthreads();
  for (int s = 128; s > 0; s >>= 1) { if (t < s) red[t] += red[t + s]; __syncthreads(); }
  if (t == 0) { rmax[(long long)b * N + row] = m; rsum[(long long)b * N + row] = red[0]; }
}

// one column per thread -> fully coalesced row sweeps, online softmax
__global__ __launch_bounds__(256)
void col_stats_kernel(const float* __restrict__ A, float* __restrict__ cmax,
                      float* __restrict__ csum, int N)
{
  const int col = blockIdx.x * 256 + threadIdx.x, b = blockIdx.y;
  if (col >= N) return;
  const float* Ab = A + (long long)b * N * N;
  float m = -3.4e38f, s = 0.f;
  for (int r = 0; r < N; ++r) {
    float x = Ab[(long long)r * N + col];
    float mn = fmaxf(m, x);
    s = s * __expf(m - mn) + __expf(x - mn);
    m = mn;
  }
  cmax[(long long)b * N + col] = m;
  csum[(long long)b * N + col] = s;
}

// M2[b][n][m] = exp(A[n][m]-rmax[n])/rsum[n]
__global__ __launch_bounds__(256)
void write_M2_kernel(const float* __restrict__ A, const float* __restrict__ rmax,
                     const float* __restrict__ rsum, bf16* __restrict__ M2, int N)
{
  const int b = blockIdx.y;
  long long i = (long long)blockIdx.x * 256 + threadIdx.x;
  const long long nn = (long long)N * N;
  if (i >= nn) return;
  int n = (int)(i / N);
  float mx = rmax[(long long)b * N + n], sm = rsum[(long long)b * N + n];
  M2[(long long)b * nn + i] = (bf16)(__expf(A[(long long)b * nn + i] - mx) / sm);
}

// M1[b][j][k] = exp(A[k][j]-cmax[j])/csum[j] via 32x32 LDS transpose
__global__ void write_M1_kernel(const float* __restrict__ A, const float* __restrict__ cmax,
                                const float* __restrict__ csum, bf16* __restrict__ M1, int N)
{
  __shared__ float tile[32][33];
  const int b = blockIdx.z;
  const float* Ab = A + (long long)b * N * N;
  bf16* Mb = M1 + (long long)b * N * N;
  const int k0 = blockIdx.y * 32, j0 = blockIdx.x * 32;
  const int tx = threadIdx.x, ty = threadIdx.y;
  for (int r = ty; r < 32; r += 8)
    tile[r][tx] = Ab[(long long)(k0 + r) * N + j0 + tx];
  __syncthreads();
  for (int r = ty; r < 32; r += 8) {
    int j = j0 + r;
    float mx = cmax[(long long)b * N + j], sm = csum[(long long)b * N + j];
    Mb[(long long)j * N + k0 + tx] = (bf16)(__expf(tile[tx][r] - mx) / sm);
  }
}

// --------------------- 3x3 conv as implicit WMMA GEMM ----------------------
// out[b][coOff+co][n] = sum_{tap,ci} Wr[tap][co][ci] * X[b][ci][shift(n,tap)]
__global__ __launch_bounds__(256)
void conv3x3_wmma(const bf16* __restrict__ Xg,   // [B][512][1600] bf16
                  const bf16* __restrict__ Wr,   // [9][512][512] bf16
                  float* __restrict__ Out,       // [B][1024][1600] f32 (d_out)
                  int coOff)
{
  __shared__ bf16 As[2][64][40];
  __shared__ unsigned Bs[16][260];

  const int bz = blockIdx.z;
  const unsigned short* Xb =
      reinterpret_cast<const unsigned short*>(Xg + (long long)bz * 512 * 1600);
  const int m0 = blockIdx.y * 64;
  const int n0 = blockIdx.x * 256;
  const int t = threadIdx.x, lane = t & 31, wave = t >> 5;
  const int wm = (wave >> 2) * 32, wn = (wave & 3) * 64;
  const int half = lane >> 4, l16 = lane & 15;
  const int bp = t >> 4, bseg = t & 15;

  v8f c0[4], c1[4];
  #pragma unroll
  for (int j = 0; j < 4; ++j) { v8f z = {}; c0[j] = z; c1[j] = z; }
  unsigned pb[16];

  // loop-invariant spatial decode for the halo loader (no div/mul in hot loop)
  int yb[16], xb[16];
  unsigned nvmask = 0;
  #pragma unroll
  for (int j = 0; j < 16; ++j) {
    int n = n0 + bseg * 16 + j;
    yb[j] = n / 40; xb[j] = n % 40;
    if (n < 1600) nvmask |= 1u << j;
  }

  // branch-free halo loader (select-zero), packed K-pairs
  auto loadBconv = [&](int s, unsigned out[16]) {
    const int tap = s >> 4;
    const int k0  = (s & 15) * 32;
    const int dy  = tap / 3 - 1, dx = tap % 3 - 1;
    const int dyx = dy * 40 + dx;
    const long long base = (long long)(k0 + 2 * bp) * 1600;
    #pragma unroll
    for (int j = 0; j < 16; ++j) {
      int n = n0 + bseg * 16 + j;
      int y = yb[j] + dy, x = xb[j] + dx;
      bool valid = ((nvmask >> j) & 1u) && ((unsigned)y < 40u) && ((unsigned)x < 40u);
      int src = valid ? (n + dyx) : 0;
      unsigned lo = Xb[base + src];
      unsigned hi = Xb[base + 1600 + src];
      if (!valid) { lo = 0u; hi = 0u; }
      out[j] = lo | (hi << 16);
    }
  };
  auto wptr = [&](int s) -> const bf16* {
    const int tap = s >> 4;
    const int k0  = (s & 15) * 32;
    return Wr + (long long)tap * 512 * 512 + (long long)m0 * 512 + k0;
  };

  const int steps = 9 * 16;  // 9 taps x (512/32) K-steps

  if (wave == 0)
    tdm_load_tile_a(wptr(0), (unsigned)(unsigned long long)&As[0][0][0], 512u, 512ull);
  loadBconv(0, pb);

  for (int i = 0; i < steps; ++i) {
    const int cur = i & 1;
    __syncthreads();
    {
      uint4* dst = reinterpret_cast<uint4*>(&Bs[bp][bseg * 16]);
      dst[0] = make_uint4(pb[0],  pb[1],  pb[2],  pb[3]);
      dst[1] = make_uint4(pb[4],  pb[5],  pb[6],  pb[7]);
      dst[2] = make_uint4(pb[8],  pb[9],  pb[10], pb[11]);
      dst[3] = make_uint4(pb[12], pb[13], pb[14], pb[15]);
    }
    if (i + 1 < steps) {
      if (wave == 0)
        tdm_load_tile_a(wptr(i + 1), (unsigned)(unsigned long long)&As[cur ^ 1][0][0],
                        512u, 512ull);
      loadBconv(i + 1, pb);
      if (wave == 0) __builtin_amdgcn_s_wait_tensorcnt(1);
    } else {
      if (wave == 0) __builtin_amdgcn_s_wait_tensorcnt(0);
    }
    __syncthreads();

    FragBF a0, a1, bf[4];
    #pragma unroll
    for (int v = 0; v < 8; ++v) {
      int kk = (v < 4 ? 2 * v : 16 + 2 * (v - 4)) + 8 * half;
      a0.u[v] = *reinterpret_cast<const unsigned*>(&As[cur][wm + l16][kk]);
      a1.u[v] = *reinterpret_cast<const unsigned*>(&As[cur][wm + 16 + l16][kk]);
      int p = (v < 4 ? v : v + 4) + 4 * half;
      #pragma unroll
      for (int tj = 0; tj < 4; ++tj)
        bf[tj].u[v] = Bs[p][wn + tj * 16 + l16];
    }
    #pragma unroll
    for (int tj = 0; tj < 4; ++tj) {
      c0[tj] = wmma_bf16(a0, bf[tj], c0[tj]);
      c1[tj] = wmma_bf16(a1, bf[tj], c1[tj]);
    }
  }

  #pragma unroll
  for (int ti = 0; ti < 2; ++ti) {
    #pragma unroll
    for (int tj = 0; tj < 4; ++tj) {
      v8f acc = ti == 0 ? c0[tj] : c1[tj];
      int col = n0 + wn + tj * 16 + l16;
      if (col < 1600) {
        #pragma unroll
        for (int r = 0; r < 8; ++r) {
          int row = m0 + wm + ti * 16 + 8 * half + r;
          Out[((long long)bz * 1024 + coOff + row) * 1600 + col] = acc[r];
        }
      }
    }
  }
}

// ------------------------------- host glue ---------------------------------

extern "C" void kernel_launch(void* const* d_in, const int* in_sizes, int n_in,
                              void* d_out, int out_size, void* d_ws, size_t ws_size,
                              hipStream_t stream) {
  (void)in_sizes; (void)n_in; (void)out_size; (void)ws_size;
  const float* exemplar = (const float*)d_in[0];  // [16,512,40,40]
  const float* query    = (const float*)d_in[1];  // [16,512,40,40]
  const float* w_e      = (const float*)d_in[2];  // [256,512]
  const float* w_q      = (const float*)d_in[3];  // [256,512]
  const float* w_c1     = (const float*)d_in[4];  // [512,512,3,3]
  const float* w_c2     = (const float*)d_in[5];  // [512,512,3,3]
  float* out = (float*)d_out;                     // [16,1024,40,40]

  const long long B = 16, C = 512, N = 1600, P = 256;

  // workspace carve-out (~470 MB total)
  char* ws = (char*)d_ws;
  long long off = 0;
  auto carve = [&](long long bytes) -> void* {
    off = (off + 255) & ~255LL;
    void* p = ws + off;
    off += bytes;
    return p;
  };
  bf16*  e_bf  = (bf16*)carve(B * C * N * 2);
  bf16*  q_bf  = (bf16*)carve(B * C * N * 2);
  bf16*  we_bf = (bf16*)carve(P * C * 2);
  bf16*  wq_bf = (bf16*)carve(P * C * 2);
  bf16*  wr1   = (bf16*)carve(9 * C * C * 2);
  bf16*  wr2   = (bf16*)carve(9 * C * C * 2);
  bf16*  ecT   = (bf16*)carve(B * N * P * 2);   // e_corr^T : [B][N][P]
  bf16*  qc    = (bf16*)carve(B * P * N * 2);   // q_corr   : [B][P][N]
  float* Amat  = (float*)carve(B * N * N * 4);
  float* rmax  = (float*)carve(B * N * 4);
  float* rsum  = (float*)carve(B * N * 4);
  float* cmax  = (float*)carve(B * N * 4);
  float* csum  = (float*)carve(B * N * 4);
  bf16*  M1    = (bf16*)carve(B * N * N * 2);
  bf16*  M2    = (bf16*)carve(B * N * N * 2);
  bf16*  eatt  = (bf16*)carve(B * C * N * 2);   // exemplar_att
  bf16*  qatt  = (bf16*)carve(B * C * N * 2);   // query_att

  // 1) precision conversion + weight reorder
  {
    long long n = B * C * N;
    cvt_f32_bf16<<<dim3((unsigned)((n + 255) / 256)), 256, 0, stream>>>(exemplar, e_bf, n);
    cvt_f32_bf16<<<dim3((unsigned)((n + 255) / 256)), 256, 0, stream>>>(query, q_bf, n);
    long long nw = P * C;
    cvt_f32_bf16<<<dim3((unsigned)((nw + 255) / 256)), 256, 0, stream>>>(w_e, we_bf, nw);
    cvt_f32_bf16<<<dim3((unsigned)((nw + 255) / 256)), 256, 0, stream>>>(w_q, wq_bf, nw);
    long long nr = 9 * C * C;
    reorder_w3x3<<<dim3((unsigned)((nr + 255) / 256)), 256, 0, stream>>>(w_c1, wr1);
    reorder_w3x3<<<dim3((unsigned)((nr + 255) / 256)), 256, 0, stream>>>(w_c2, wr2);
  }

  // 2) 1x1 projections: e_corr stored transposed [N,256]; q_corr normal [256,N]
  gemm_bf16_wmma<2><<<dim3(7, 4, 16), 256, 0, stream>>>(
      we_bf, e_bf, ecT, 256, 1600, 512, 0, C * N, N * P, 256);
  gemm_bf16_wmma<1><<<dim3(7, 4, 16), 256, 0, stream>>>(
      wq_bf, q_bf, qc, 256, 1600, 512, 0, C * N, P * N, 1600);

  // 3) affinity: A[b] = ecT[N,256] x qc[256,N] -> f32 [N,N]
  gemm_bf16_wmma<0><<<dim3(7, 25, 16), 256, 0, stream>>>(
      ecT, qc, Amat, 1600, 1600, 256, N * P, P * N, N * N, 1600);

  // 4) softmax stats (rows and columns) + normalized bf16 operands M2, M1
  row_stats_kernel<<<dim3(1600, 16), 256, 0, stream>>>(Amat, rmax, rsum, 1600);
  col_stats_kernel<<<dim3(7, 16), 256, 0, stream>>>(Amat, cmax, csum, 1600);
  write_M2_kernel<<<dim3(10000, 16), 256, 0, stream>>>(Amat, rmax, rsum, M2, 1600);
  write_M1_kernel<<<dim3(50, 50, 16), dim3(32, 8), 0, stream>>>(Amat, cmax, csum, M1, 1600);

  // 5) attention matmuls: exemplar_att = q x M2 ; query_att = e x M1
  gemm_bf16_wmma<1><<<dim3(7, 8, 16), 256, 0, stream>>>(
      q_bf, M2, eatt, 512, 1600, 1600, C * N, N * N, C * N, 1600);
  gemm_bf16_wmma<1><<<dim3(7, 8, 16), 256, 0, stream>>>(
      e_bf, M1, qatt, 512, 1600, 1600, C * N, N * N, C * N, 1600);

  // 6) 3x3 SAME convs (implicit WMMA GEMM) -> concatenated f32 output
  conv3x3_wmma<<<dim3(7, 8, 16), 256, 0, stream>>>(eatt, wr1, out, 0);    // ch 0..511
  conv3x3_wmma<<<dim3(7, 8, 16), 256, 0, stream>>>(qatt, wr2, out, 512);  // ch 512..1023
}
